// Layer_scale_init_BlockD8_87222195848233
// MI455X (gfx1250) — matrix operand order repaired and checked by reading, not compiled, not run on hardware
//
#include <hip/hip_runtime.h>

// ---------------------------------------------------------------------------
// Transformer block (LayerScale, D8-decomposed streams) for MI455X / gfx1250.
// Compute-bound (~95 GFLOP vs ~400MB traffic @23.3TB/s) -> everything routed
// through v_wmma_f32_16x16x32_bf16, flash attention keeps S in registers/LDS.
// ---------------------------------------------------------------------------

#define DIMC   768
#define HEADS  12
#define BATCH  16
#define SEQ    1024
#define TOK    (BATCH*SEQ)      // 16384 tokens
#define HD     64               // head dim

typedef __attribute__((ext_vector_type(16))) __bf16 v16bf;
typedef __attribute__((ext_vector_type(8)))  float  v8f;
typedef __attribute__((ext_vector_type(4)))  int    v4i;

union FragB { uint4 q[2]; v16bf v; };

__device__ __forceinline__ unsigned short f2bf(float f) {
  unsigned u = __builtin_bit_cast(unsigned, f);
  unsigned r = u + 0x7FFFu + ((u >> 16) & 1u);     // round-to-nearest-even
  return (unsigned short)(r >> 16);
}

// ---- global -> LDS copy: async-to-LDS when toolchain exposes it -----------
#ifndef __has_builtin
#define __has_builtin(x) 0
#endif
#if __has_builtin(__builtin_amdgcn_global_load_async_to_lds_b128)
#define HAVE_ASYNC_CP 1
#else
#define HAVE_ASYNC_CP 0
#endif

typedef __attribute__((address_space(1))) v4i* as1_v4i;  // global int4*
typedef __attribute__((address_space(3))) v4i* as3_v4i;  // LDS int4*

__device__ __forceinline__ void cp16_g2s(const unsigned short* __restrict__ g,
                                         unsigned short* __restrict__ s) {
#if HAVE_ASYNC_CP
  __builtin_amdgcn_global_load_async_to_lds_b128(
      (as1_v4i)(unsigned long long)(const void*)g,
      (as3_v4i)(unsigned)(unsigned long long)(const void*)s,
      0, 0);
#else
  *(uint4*)s = *(const uint4*)g;
#endif
}
__device__ __forceinline__ void cp_wait() {
#if HAVE_ASYNC_CP
#if __has_builtin(__builtin_amdgcn_s_wait_asynccnt)
  __builtin_amdgcn_s_wait_asynccnt(0);
#else
  asm volatile("s_wait_asynccnt 0" ::: "memory");
#endif
#endif
}

// ---------------------------------------------------------------------------
// f32 -> bf16 conversion (weights, once per call)
// ---------------------------------------------------------------------------
__global__ void cvt_bf16_kernel(const float* __restrict__ src,
                                unsigned short* __restrict__ dst, long n) {
  long i = (long)blockIdx.x * blockDim.x + threadIdx.x;
  if (i < n) dst[i] = f2bf(src[i]);
}

// ---------------------------------------------------------------------------
// Fused joint layernorm over the 5 streams + affine + bf16 pack.
// One block per token. Stats via LDS float atomics (ds_add_f32).
// Output layout (dense-5): [A1 T*96][A2][B1][B2][E 2T*192]
// ---------------------------------------------------------------------------
__global__ __launch_bounds__(256) void ln_pack_kernel(
    const float* __restrict__ x0, const float* __restrict__ x1,
    const float* __restrict__ x2, const float* __restrict__ x3,
    const float* __restrict__ xE,
    const float* __restrict__ g0, const float* __restrict__ g1,
    const float* __restrict__ g2, const float* __restrict__ g3,
    const float* __restrict__ gE, const float* __restrict__ beta,
    unsigned short* __restrict__ out) {
  __shared__ float ssum[6], ssq[6], smean[6], sinv;
  long t = blockIdx.x;
  int tid = threadIdx.x;
  if (tid < 6) { ssum[tid] = 0.f; ssq[tid] = 0.f; }
  __syncthreads();
  float vals[3]; int segs[3], chs[3];
#pragma unroll
  for (int k = 0; k < 3; k++) {
    int c = tid + k * 256;
    float v; int seg, ch;
    if (c < 384) {
      seg = c / 96; ch = c % 96;
      const float* xs = (seg == 0) ? x0 : (seg == 1) ? x1 : (seg == 2) ? x2 : x3;
      v = xs[t * 96 + ch];
    } else {
      int e = c - 384; seg = 4 + e / 192; ch = e % 192;
      v = xE[t * 384 + e];
    }
    vals[k] = v; segs[k] = seg; chs[k] = ch;
    atomicAdd(&ssum[seg], v);
    atomicAdd(&ssq[seg], v * v);
  }
  __syncthreads();
  if (tid == 0) {
    float m[6], va[6];
    for (int s = 0; s < 6; s++) {
      float n = (s < 4) ? 96.f : 192.f;
      m[s] = ssum[s] / n;
      va[s] = ssq[s] / n - m[s] * m[s];
      smean[s] = m[s];
    }
    float vtot = va[0] + va[1] + va[2] + va[3] + 0.5f * (va[4] + va[5]) + 1e-5f;
    sinv = 1.f / (0.3535533905932738f * sqrtf(vtot));
  }
  __syncthreads();
  float inv = sinv;
#pragma unroll
  for (int k = 0; k < 3; k++) {
    int c = tid + k * 256;
    int seg = segs[k], ch = chs[k];
    float y = (vals[k] - smean[seg]) * inv;
    long oidx;
    if (seg < 4) {
      const float* gs = (seg == 0) ? g0 : (seg == 1) ? g1 : (seg == 2) ? g2 : g3;
      y *= gs[ch];
      if (seg == 0 && beta) y += beta[ch];
      oidx = (long)seg * TOK * 96 + t * 96 + ch;
    } else {
      y *= gE[ch];
      oidx = 4L * TOK * 96 + t * 384 + (c - 384);
    }
    out[oidx] = f2bf(y);
  }
}

// ---------------------------------------------------------------------------
// bf16 WMMA GEMM: C[M,N] = A[M,K] @ W[N,K]^T (+bias) (opt. tanh-GELU)
// Block 256 thr = 8 waves; tile 128x96x32; wave tile 32x48 (2x3 WMMA).
// LDS: K-contiguous, stride 48 elems (96B) -> b128-aligned fragment loads.
// ---------------------------------------------------------------------------
#define BM 128
#define BN 96
#define BK 32
#define LDSK 48

__global__ __launch_bounds__(256) void gemm_bf16_wmma(
    const unsigned short* __restrict__ A, const unsigned short* __restrict__ W,
    const float* __restrict__ bias, void* __restrict__ Cout,
    int M, int N, int K, int out_f32, int do_gelu) {
  __shared__ __align__(16) unsigned short As[BM * LDSK];
  __shared__ __align__(16) unsigned short Bs[BN * LDSK];
  const int tid = threadIdx.x;
  const int lane = tid & 31, wid = tid >> 5;
  const int r = lane & 15, h = lane >> 4;
  const int wm = wid & 3, wn = wid >> 2;
  const int m0 = blockIdx.y * BM;
  const int n0 = blockIdx.x * BN;

  v8f zero = {0.f, 0.f, 0.f, 0.f, 0.f, 0.f, 0.f, 0.f};
  v8f acc[2][3];
  for (int i = 0; i < 2; i++)
    for (int j = 0; j < 3; j++) acc[i][j] = zero;

  for (int kt = 0; kt < K; kt += BK) {
    __syncthreads();
    {  // A tile 128x32: 512 chunks of 8 bf16
      int c = tid;
#pragma unroll
      for (int it = 0; it < 2; ++it, c += 256) {
        int row = c >> 2, kc = (c & 3) * 8;
        cp16_g2s(A + (long)(m0 + row) * K + kt + kc, &As[row * LDSK + kc]);
      }
    }
    {  // B tile 96x32: 384 chunks
      int c = tid;
#pragma unroll
      for (int it = 0; it < 2; ++it, c += 256) {
        if (c < (BN * BK / 8)) {
          int row = c >> 2, kc = (c & 3) * 8;
          cp16_g2s(W + (long)(n0 + row) * K + kt + kc, &Bs[row * LDSK + kc]);
        }
      }
    }
    cp_wait();
    __syncthreads();

    FragB a[2], b[3];
#pragma unroll
    for (int mi = 0; mi < 2; mi++) {
      int rowa = wm * 32 + mi * 16 + r;
      a[mi].q[0] = *(const uint4*)&As[rowa * LDSK + h * 8];
      a[mi].q[1] = *(const uint4*)&As[rowa * LDSK + 16 + h * 8];
    }
#pragma unroll
    for (int ni = 0; ni < 3; ni++) {
      int colb = wn * 48 + ni * 16 + r;
      b[ni].q[0] = *(const uint4*)&Bs[colb * LDSK + h * 8];
      b[ni].q[1] = *(const uint4*)&Bs[colb * LDSK + 16 + h * 8];
    }
#pragma unroll
    for (int mi = 0; mi < 2; mi++)
#pragma unroll
      for (int ni = 0; ni < 3; ni++)
        acc[mi][ni] = __builtin_amdgcn_wmma_f32_16x16x32_bf16(
            false, a[mi].v, false, b[ni].v, (short)0, acc[mi][ni], false, false);
  }

#pragma unroll
  for (int mi = 0; mi < 2; mi++)
#pragma unroll
    for (int ni = 0; ni < 3; ni++) {
      int col = n0 + wn * 48 + ni * 16 + r;
      float bv = bias ? bias[col] : 0.f;
#pragma unroll
      for (int v = 0; v < 8; v++) {
        int row = m0 + wm * 32 + mi * 16 + h * 8 + v;
        float val = acc[mi][ni][v] + bv;
        if (do_gelu) {
          float x3 = val * val * val;
          val = 0.5f * val *
                (1.f + tanhf(0.7978845608028654f * (val + 0.044715f * x3)));
        }
        long idx = (long)row * N + col;
        if (out_f32) ((float*)Cout)[idx] = val;
        else ((unsigned short*)Cout)[idx] = f2bf(val);
      }
    }
}

// ---------------------------------------------------------------------------
// Gather per-stream qkv GEMM outputs into Q/K/V [B*H][N][64] bf16.
// ---------------------------------------------------------------------------
__global__ void assemble_qkv(const unsigned short* __restrict__ Y,
                             unsigned short* __restrict__ Q,
                             unsigned short* __restrict__ Kp,
                             unsigned short* __restrict__ V) {
  long idx = (long)blockIdx.x * blockDim.x + threadIdx.x;
  long total = (long)BATCH * HEADS * SEQ * HD;
  if (idx >= total) return;
  int d = (int)(idx & 63);
  long rem = idx >> 6;
  int n = (int)(rem & 1023);
  long bh = rem >> 10;
  int h = (int)(bh % 12);
  long b = bh / 12;
  long t = b * SEQ + n;
  const unsigned short* YE = Y + 4L * TOK * 288;
  unsigned short qv, kv, vv;
  if (d < 32) {
    int si = d >> 3, inner = d & 7;
    const unsigned short* ys = Y + (long)si * TOK * 288 + t * 288;
    int col = h * 8 + inner;
    qv = ys[col]; kv = ys[96 + col]; vv = ys[192 + col];
  } else {
    int s = (d - 32) >> 4, dd = (d - 32) & 15;
    const unsigned short* ye = YE + (t * 2 + s) * 576;
    int col = h * 16 + dd;
    qv = ye[col]; kv = ye[192 + col]; vv = ye[384 + col];
  }
  Q[idx] = qv; Kp[idx] = kv; V[idx] = vv;
}

// ---------------------------------------------------------------------------
// Flash attention per (head, 64-query tile); 4 waves, online softmax.
// ---------------------------------------------------------------------------
#define AT_LD 72

__global__ __launch_bounds__(128) void attn_kernel(
    const unsigned short* __restrict__ Q, const unsigned short* __restrict__ Kp,
    const unsigned short* __restrict__ V, unsigned short* __restrict__ O) {
  __shared__ __align__(16) unsigned short Qs[64 * AT_LD];
  __shared__ __align__(16) unsigned short Ks[64 * AT_LD];
  __shared__ __align__(16) unsigned short Vs[HD * AT_LD];
  __shared__ __align__(16) unsigned short Ps[4][16 * AT_LD];

  int tid = threadIdx.x, lane = tid & 31, w = tid >> 5;
  int r = lane & 15, h = lane >> 4;
  long bh = blockIdx.y;
  int q0 = blockIdx.x * 64;
  const long base = bh * SEQ * HD;

  for (int c = tid; c < 64 * HD / 8; c += 128) {
    int row = c >> 3, k8 = (c & 7) * 8;
    *(uint4*)&Qs[row * AT_LD + k8] = *(const uint4*)&Q[base + (long)(q0 + row) * HD + k8];
  }
  v8f zero = {0.f, 0.f, 0.f, 0.f, 0.f, 0.f, 0.f, 0.f};
  v8f o[4];
  for (int i = 0; i < 4; i++) o[i] = zero;
  float mrow[8], lrow[8];
  for (int v = 0; v < 8; v++) { mrow[v] = -1e30f; lrow[v] = 0.f; }

  for (int kv0 = 0; kv0 < SEQ; kv0 += 64) {
    __syncthreads();
    for (int c = tid; c < 64 * HD / 8; c += 128) {  // K tile row-major
      int row = c >> 3, k8 = (c & 7) * 8;
      *(uint4*)&Ks[row * AT_LD + k8] = *(const uint4*)&Kp[base + (long)(kv0 + row) * HD + k8];
    }
    for (int c = tid; c < 64 * HD / 8; c += 128) {  // V tile transposed
      int j = c >> 3, d0 = (c & 7) * 8;
      union { uint4 u; unsigned short s[8]; } tmp;
      tmp.u = *(const uint4*)&V[base + (long)(kv0 + j) * HD + d0];
#pragma unroll
      for (int e = 0; e < 8; e++) Vs[(d0 + e) * AT_LD + j] = tmp.s[e];
    }
    __syncthreads();

    v8f sc[4];
    for (int i = 0; i < 4; i++) sc[i] = zero;
#pragma unroll
    for (int ks = 0; ks < HD; ks += 32) {
      FragB aq;
      int rowa = w * 16 + r;
      aq.q[0] = *(const uint4*)&Qs[rowa * AT_LD + ks + h * 8];
      aq.q[1] = *(const uint4*)&Qs[rowa * AT_LD + ks + 16 + h * 8];
#pragma unroll
      for (int nt = 0; nt < 4; nt++) {
        FragB bk;
        int colb = nt * 16 + r;
        bk.q[0] = *(const uint4*)&Ks[colb * AT_LD + ks + h * 8];
        bk.q[1] = *(const uint4*)&Ks[colb * AT_LD + ks + 16 + h * 8];
        sc[nt] = __builtin_amdgcn_wmma_f32_16x16x32_bf16(
            false, aq.v, false, bk.v, (short)0, sc[nt], false, false);
      }
    }
#pragma unroll
    for (int nt = 0; nt < 4; nt++)
#pragma unroll
      for (int v = 0; v < 8; v++) sc[nt][v] *= 0.125f;  // 1/sqrt(64)

#pragma unroll
    for (int v = 0; v < 8; v++) {
      float mx = fmaxf(fmaxf(sc[0][v], sc[1][v]), fmaxf(sc[2][v], sc[3][v]));
#pragma unroll
      for (int m_ = 1; m_ < 16; m_ <<= 1) mx = fmaxf(mx, __shfl_xor(mx, m_));
      float mnew = fmaxf(mrow[v], mx);
      float alpha = __expf(mrow[v] - mnew);
      float s = 0.f;
#pragma unroll
      for (int nt = 0; nt < 4; nt++) {
        float p = __expf(sc[nt][v] - mnew);
        sc[nt][v] = p; s += p;
      }
#pragma unroll
      for (int m_ = 1; m_ < 16; m_ <<= 1) s += __shfl_xor(s, m_);
      lrow[v] = lrow[v] * alpha + s;
      mrow[v] = mnew;
#pragma unroll
      for (int nt = 0; nt < 4; nt++) o[nt][v] *= alpha;
    }
#pragma unroll
    for (int nt = 0; nt < 4; nt++)  // C-layout -> A-layout via LDS
#pragma unroll
      for (int v = 0; v < 8; v++)
        Ps[w][(v + h * 8) * AT_LD + nt * 16 + r] = f2bf(sc[nt][v]);
    __syncthreads();
#pragma unroll
    for (int ks = 0; ks < 64; ks += 32) {  // O += P @ V
      FragB ap;
      ap.q[0] = *(const uint4*)&Ps[w][r * AT_LD + ks + h * 8];
      ap.q[1] = *(const uint4*)&Ps[w][r * AT_LD + ks + 16 + h * 8];
#pragma unroll
      for (int nt = 0; nt < 4; nt++) {
        FragB bv;
        int colb = nt * 16 + r;
        bv.q[0] = *(const uint4*)&Vs[colb * AT_LD + ks + h * 8];
        bv.q[1] = *(const uint4*)&Vs[colb * AT_LD + ks + 16 + h * 8];
        o[nt] = __builtin_amdgcn_wmma_f32_16x16x32_bf16(
            false, ap.v, false, bv.v, (short)0, o[nt], false, false);
      }
    }
  }
#pragma unroll
  for (int nt = 0; nt < 4; nt++)
#pragma unroll
    for (int v = 0; v < 8; v++) {
      float val = o[nt][v] / lrow[v];
      int row = q0 + w * 16 + h * 8 + v;
      O[base + (long)row * HD + nt * 16 + r] = f2bf(val);
    }
}

// ---------------------------------------------------------------------------
// O [B*H][N][64] -> dense-5 packed bf16 (proj GEMM input)
// ---------------------------------------------------------------------------
__global__ void disassemble_o(const unsigned short* __restrict__ O,
                              unsigned short* __restrict__ X2) {
  long i = (long)blockIdx.x * blockDim.x + threadIdx.x;
  if (i >= (long)TOK * DIMC) return;
  long t; int h, d;
  if (i < 4L * TOK * 96) {
    int si = (int)(i / ((long)TOK * 96));
    long rem = i - (long)si * TOK * 96;
    t = rem / 96; int c = (int)(rem % 96);
    h = c >> 3; d = si * 8 + (c & 7);
  } else {
    long rem = i - 4L * TOK * 96;
    t = rem / 384; int c = (int)(rem % 384);
    int s = c / 192, cc = c % 192;
    h = cc >> 4; d = 32 + s * 16 + (cc & 15);
  }
  long b = t >> 10; int n = (int)(t & 1023);
  X2[i] = O[((b * 12 + h) * (long)SEQ + n) * HD + d];
}

// ---------------------------------------------------------------------------
// out = x + gamma * y  (LayerScale + residual); x may alias out.
// ---------------------------------------------------------------------------
__global__ void residual_g(const float* __restrict__ x0, const float* __restrict__ x1,
                           const float* __restrict__ x2, const float* __restrict__ x3,
                           const float* __restrict__ xE,
                           const float* __restrict__ g0, const float* __restrict__ g1,
                           const float* __restrict__ g2, const float* __restrict__ g3,
                           const float* __restrict__ gE,
                           const float* __restrict__ y, float* __restrict__ out) {
  long i = (long)blockIdx.x * blockDim.x + threadIdx.x;
  if (i >= (long)TOK * DIMC) return;
  float xv, gv;
  if (i < 4L * TOK * 96) {
    int si = (int)(i / ((long)TOK * 96));
    long rem = i - (long)si * TOK * 96;
    int ch = (int)(rem % 96);
    const float* xs = (si == 0) ? x0 : (si == 1) ? x1 : (si == 2) ? x2 : x3;
    const float* gs = (si == 0) ? g0 : (si == 1) ? g1 : (si == 2) ? g2 : g3;
    xv = xs[rem]; gv = gs[ch];
  } else {
    long rem = i - 4L * TOK * 96;
    xv = xE[rem]; gv = gE[(int)(rem % 192)];
  }
  out[i] = xv + gv * y[i];
}

// ---------------------------------------------------------------------------
// Host launcher. Input order = jax pytree flatten (dicts sorted by key,
// None leaves dropped): x_A1..x_E, then fc1, fc2, g1, g2, norm1, norm2,
// proj, qkv (inner keys sorted A1,A2,B1,B2,E,bA1 / aA1..aE,beta).
// ---------------------------------------------------------------------------
enum {
  IN_XA1 = 0, IN_XA2, IN_XB1, IN_XB2, IN_XE,
  FC1_A1 = 5, FC1_A2, FC1_B1, FC1_B2, FC1_E, FC1_BIAS,
  FC2_A1 = 11, FC2_A2, FC2_B1, FC2_B2, FC2_E, FC2_BIAS,
  G1_A1 = 17, G1_A2, G1_B1, G1_B2, G1_E,
  G2_A1 = 22, G2_A2, G2_B1, G2_B2, G2_E,
  N1_A1 = 27, N1_A2, N1_B1, N1_B2, N1_E, N1_BETA,
  N2_A1 = 33, N2_A2, N2_B1, N2_B2, N2_E, N2_BETA,
  PJ_A1 = 39, PJ_A2, PJ_B1, PJ_B2, PJ_E, PJ_BIAS,
  QK_A1 = 45, QK_A2, QK_B1, QK_B2, QK_E
};

extern "C" void kernel_launch(void* const* d_in, const int* in_sizes, int n_in,
                              void* d_out, int out_size, void* d_ws, size_t ws_size,
                              hipStream_t stream) {
  auto F = [&](int i) { return (const float*)d_in[i]; };

  char* ws = (char*)d_ws;
  size_t off = 0;
  auto alloc = [&](size_t bytes) -> void* {
    void* p = ws + off; off += (bytes + 255) & ~(size_t)255; return p;
  };
  unsigned short* WQ = (unsigned short*)alloc((size_t)221184 * 2);
  unsigned short* WP = (unsigned short*)alloc((size_t)73728 * 2);
  unsigned short* W1 = (unsigned short*)alloc((size_t)294912 * 2);
  unsigned short* W2 = (unsigned short*)alloc((size_t)294912 * 2);
  unsigned short* RX = (unsigned short*)alloc((size_t)TOK * 768 * 2);   // LN out / X2 / LN2 out
  unsigned short* RY = (unsigned short*)alloc((size_t)TOK * 3072 * 2);  // qkv out, later H1
  unsigned short* Qp = (unsigned short*)alloc((size_t)TOK * 768 * 2);
  unsigned short* Kp = (unsigned short*)alloc((size_t)TOK * 768 * 2);
  unsigned short* Vp = (unsigned short*)alloc((size_t)TOK * 768 * 2);
  unsigned short* Ob = (unsigned short*)alloc((size_t)TOK * 768 * 2);
  float*          RF = (float*)alloc((size_t)TOK * 768 * 4);            // proj out, later H2

  auto cvt = [&](int idx, unsigned short* dst, long n) {
    cvt_bf16_kernel<<<(unsigned)((n + 255) / 256), 256, 0, stream>>>(F(idx), dst, n);
  };
  // qkv weights (A blocks 288x96, E 576x192)
  cvt(QK_A1, WQ + 0, 27648); cvt(QK_A2, WQ + 27648, 27648);
  cvt(QK_B1, WQ + 55296, 27648); cvt(QK_B2, WQ + 82944, 27648);
  cvt(QK_E, WQ + 110592, 110592);
  // proj (96x96, E 192x192)
  cvt(PJ_A1, WP + 0, 9216); cvt(PJ_A2, WP + 9216, 9216);
  cvt(PJ_B1, WP + 18432, 9216); cvt(PJ_B2, WP + 27648, 9216);
  cvt(PJ_E, WP + 36864, 36864);
  // fc1 (384x96, E 768x192)
  cvt(FC1_A1, W1 + 0, 36864); cvt(FC1_A2, W1 + 36864, 36864);
  cvt(FC1_B1, W1 + 73728, 36864); cvt(FC1_B2, W1 + 110592, 36864);
  cvt(FC1_E, W1 + 147456, 147456);
  // fc2 (96x384, E 192x768)
  cvt(FC2_A1, W2 + 0, 36864); cvt(FC2_A2, W2 + 36864, 36864);
  cvt(FC2_B1, W2 + 73728, 36864); cvt(FC2_B2, W2 + 110592, 36864);
  cvt(FC2_E, W2 + 147456, 147456);

  auto gemm = [&](const unsigned short* A, const unsigned short* W,
                  const float* bias, void* C, int M, int N, int K,
                  int out_f32, int gelu) {
    dim3 g(N / BN, M / BM);
    gemm_bf16_wmma<<<g, 256, 0, stream>>>(A, W, bias, C, M, N, K, out_f32, gelu);
  };
  const unsigned EW = (unsigned)(((long)TOK * 768 + 255) / 256);  // elementwise grid

  // ---- LN1 -> RX (bf16 dense-5) ----
  ln_pack_kernel<<<TOK, 256, 0, stream>>>(
      F(IN_XA1), F(IN_XA2), F(IN_XB1), F(IN_XB2), F(IN_XE),
      F(N1_A1), F(N1_A2), F(N1_B1), F(N1_B2), F(N1_E), F(N1_BETA), RX);

  // ---- qkv GEMMs -> RY ----
  for (int si = 0; si < 4; si++)
    gemm(RX + (size_t)si * TOK * 96, WQ + si * 27648, nullptr,
         RY + (size_t)si * TOK * 288, TOK, 288, 96, 0, 0);
  gemm(RX + 4L * TOK * 96, WQ + 110592, nullptr,
       RY + 4L * TOK * 288, 2 * TOK, 576, 192, 0, 0);

  // ---- assemble heads, flash attention, disassemble ----
  assemble_qkv<<<EW, 256, 0, stream>>>(RY, Qp, Kp, Vp);
  attn_kernel<<<dim3(SEQ / 64, BATCH * HEADS), 128, 0, stream>>>(Qp, Kp, Vp, Ob);
  disassemble_o<<<EW, 256, 0, stream>>>(Ob, RX);

  // ---- proj GEMMs -> RF (f32, dense-5) ----
  for (int si = 0; si < 4; si++)
    gemm(RX + (size_t)si * TOK * 96, WP + si * 9216,
         si == 0 ? F(PJ_BIAS) : nullptr,
         RF + (size_t)si * TOK * 96, TOK, 96, 96, 1, 0);
  gemm(RX + 4L * TOK * 96, WP + 36864, nullptr,
       RF + 4L * TOK * 96, 2 * TOK, 192, 192, 1, 0);

  // ---- residual 1: d_out = x + g1 * proj ----
  float* ob = (float*)d_out;
  residual_g<<<EW, 256, 0, stream>>>(
      F(IN_XA1), F(IN_XA2), F(IN_XB1), F(IN_XB2), F(IN_XE),
      F(G1_A1), F(G1_A2), F(G1_B1), F(G1_B2), F(G1_E), RF, ob);

  // ---- LN2 (reads d_out) -> RX ----
  ln_pack_kernel<<<TOK, 256, 0, stream>>>(
      ob, ob + (long)TOK * 96, ob + 2L * TOK * 96, ob + 3L * TOK * 96,
      ob + 4L * TOK * 96,
      F(N2_A1), F(N2_A2), F(N2_B1), F(N2_B2), F(N2_E), F(N2_BETA), RX);

  // ---- fc1 (+bias A1, GELU) -> RY bf16 ----
  for (int si = 0; si < 4; si++)
    gemm(RX + (size_t)si * TOK * 96, W1 + si * 36864,
         si == 0 ? F(FC1_BIAS) : nullptr,
         RY + (size_t)si * TOK * 384, TOK, 384, 96, 0, 1);
  gemm(RX + 4L * TOK * 96, W1 + 147456, nullptr,
       RY + 4L * TOK * 384, 2 * TOK, 768, 192, 0, 1);

  // ---- fc2 -> RF f32 ----
  for (int si = 0; si < 4; si++)
    gemm(RY + (size_t)si * TOK * 384, W2 + si * 36864,
         si == 0 ? F(FC2_BIAS) : nullptr,
         RF + (size_t)si * TOK * 96, TOK, 96, 384, 1, 0);
  gemm(RY + 4L * TOK * 384, W2 + 147456, nullptr,
       RF + 4L * TOK * 96, 2 * TOK, 192, 768, 1, 0);

  // ---- residual 2: d_out += g2 * h ----
  residual_g<<<EW, 256, 0, stream>>>(
      ob, ob + (long)TOK * 96, ob + 2L * TOK * 96, ob + 3L * TOK * 96,
      ob + 4L * TOK * 96,
      F(G2_A1), F(G2_A2), F(G2_B1), F(G2_B2), F(G2_E), RF, ob);
}
